// MegalodonAttention_1503238554307
// MI455X (gfx1250) — matrix operand order repaired and checked by reading, not compiled, not run on hardware
//
#include <hip/hip_runtime.h>
#include <hip/hip_bf16.h>
#include <stdint.h>

typedef __attribute__((ext_vector_type(16))) _Float16 v16h;
typedef __attribute__((ext_vector_type(8)))  _Float16 v8h;
typedef __attribute__((ext_vector_type(8)))  float    v8f;

#define B_ 4
#define T_ 4096
#define H_ 16
#define DH_ 128
#define DV_ 128
#define CHUNK_ 1024
#define NEGINF (-1.0e30f)
// ln(10000)/64
#define ROPE_SCALE 0.14391156831212787f

union AFrag { v16h v; v8h h[2]; };
union HPack { _Float16 h[2]; uint32_t u; };

// ---------------------------------------------------------------------------
// Pre-pass: RoPE(Q), RoPE(K) -> f16; V -> f16.  Pure bandwidth (~576MB moved).
// ---------------------------------------------------------------------------
__launch_bounds__(256)
__global__ void rope_cast_kernel(const float* __restrict__ Q,
                                 const float* __restrict__ K,
                                 const float* __restrict__ V,
                                 const int* __restrict__ start_index,
                                 _Float16* __restrict__ Qr,
                                 _Float16* __restrict__ Kr,
                                 _Float16* __restrict__ Vh)
{
    const int start = start_index[0];
    const size_t stride = (size_t)gridDim.x * blockDim.x;
    const size_t tid0   = (size_t)blockIdx.x * blockDim.x + threadIdx.x;

    const size_t npairs = (size_t)B_ * T_ * H_ * (DH_ / 2);
    for (size_t p = tid0; p < npairs; p += stride) {
        size_t row = p >> 6;                 // (b*T + t)*H + h
        int d = (int)(p & 63);
        int t = (int)((row / H_) % T_);
        float ang = (float)(start + t) * __expf(-ROPE_SCALE * (float)d);
        float sn, cs;
        __sincosf(ang, &sn, &cs);
        size_t base = row * DH_;
        float q1 = Q[base + d], q2 = Q[base + d + 64];
        Qr[base + d]      = (_Float16)(q1 * cs - q2 * sn);
        Qr[base + d + 64] = (_Float16)(q2 * cs + q1 * sn);
        float k1 = K[base + d], k2 = K[base + d + 64];
        Kr[base + d]      = (_Float16)(k1 * cs - k2 * sn);
        Kr[base + d + 64] = (_Float16)(k2 * cs + k1 * sn);
    }
    const size_t nel = (size_t)B_ * T_ * H_ * DV_;
    for (size_t e = tid0; e < nel; e += stride)
        Vh[e] = (_Float16)V[e];
}

// gfx1250 async global->LDS copy (ASYNCcnt-tracked), saddr + 32-bit voffset form
__device__ __forceinline__ void async_b128(uint32_t lds_addr, uint32_t voff,
                                           unsigned long long saddr)
{
    asm volatile("global_load_async_to_lds_b128 %0, %1, %2"
                 :: "v"(lds_addr), "v"(voff), "s"(saddr) : "memory");
}

// ---------------------------------------------------------------------------
// Fast path: flash attention on pre-RoPE'd f16 tensors.
// ---------------------------------------------------------------------------
__launch_bounds__(256)
__global__ void megalodon_attn_f16_kernel(const _Float16* __restrict__ Qr,
                                          const _Float16* __restrict__ Kr,
                                          const _Float16* __restrict__ Vh,
                                          float* __restrict__ O)
{
    __shared__ __align__(64) _Float16 smem[24576];
    _Float16* sQ  = smem;           // 128*128
    _Float16* sK  = smem;           // 64*128
    _Float16* sVT = smem + 8192;    // 128*64  ([dv][key])
    _Float16* sP  = smem + 16384;   // 8 * 16*64

    const int tid  = threadIdx.x;
    const int lane = tid & 31;
    const int wave = tid >> 5;
    const int col  = lane & 15;
    const int hi   = (lane >> 4) & 1;

    // Scalar (provably uniform) wave index: makes all causal-guard conditions
    // pure SGPR expressions -> s_cmp/s_cbranch, no exec manipulation.
    const int wave_s = __builtin_amdgcn_readfirstlane(wave);

    const int qb_per_bh = T_ / 128;          // 32
    int idx = blockIdx.x;
    const int b  = idx / (H_ * qb_per_bh);
    idx -= b * H_ * qb_per_bh;
    const int h  = idx / qb_per_bh;
    const int qb = idx - h * qb_per_bh;

    const int qbase       = qb * 128;
    const int chunk_start = (qbase / CHUNK_) * CHUNK_;
    const int qrel        = qbase - chunk_start;
    const int num_tiles   = (qrel + 128) >> 6;

    const uint32_t lds_q0 = (uint32_t)(uintptr_t)sQ;
    const uint32_t lds_k0 = (uint32_t)(uintptr_t)sK;

    // ---- async stage Q tile (128 rows x 256B = 2048 16B chunks, 8/thread) ----
    {
        const _Float16* qp = Qr + ((size_t)((b * T_ + qbase) * H_ + h)) * DH_;
        unsigned long long sq = (unsigned long long)(uintptr_t)qp;
        #pragma unroll
        for (int i = 0; i < 8; ++i) {
            int ch = i * 256 + tid;
            uint32_t voff = (uint32_t)((ch >> 4) * (H_ * DH_ * 2) + (ch & 15) * 16);
            async_b128(lds_q0 + (uint32_t)ch * 16, voff, sq);
        }
    }
    asm volatile("s_wait_asynccnt 0" ::: "memory");
    __syncthreads();

    // ---- per-wave Q A-fragments ----
    AFrag qf[4];
    {
        const int qrow = wave_s * 16 + col;
        const int s8   = hi ? 16 : 0;
        #pragma unroll
        for (int kk = 0; kk < 4; ++kk) {
            const _Float16* base = sQ + qrow * DH_ + kk * 32;
            qf[kk].h[0] = *(const v8h*)(base + s8);
            qf[kk].h[1] = *(const v8h*)(base + 8 + s8);
        }
    }
    asm volatile("s_wait_dscnt 0" ::: "memory");   // Q reads done before overwrite

    v8f acc[8];
    for (int n = 0; n < 8; ++n) acc[n] = (v8f){};
    float m_i[8], l_i[8];
    for (int r = 0; r < 8; ++r) { m_i[r] = NEGINF; l_i[r] = 0.0f; }

    const int qfirst = qbase + wave_s * 16;        // scalar: wave's first q row
    const int qlast  = qfirst + 15;                // scalar: wave's last q row

    for (int kt = 0; kt < num_tiles; ++kt) {
        const int kstart = chunk_start + kt * 64;
        __syncthreads();   // previous-iteration LDS reads complete

        // ---- async stage K tile (64 rows x 256B = 1024 chunks, 4/thread) ----
        {
            const _Float16* kp = Kr + ((size_t)((b * T_ + kstart) * H_ + h)) * DH_;
            unsigned long long sk = (unsigned long long)(uintptr_t)kp;
            #pragma unroll
            for (int i = 0; i < 4; ++i) {
                int ch = i * 256 + tid;
                uint32_t voff = (uint32_t)((ch >> 4) * (H_ * DH_ * 2) + (ch & 15) * 16);
                async_b128(lds_k0 + (uint32_t)ch * 16, voff, sk);
            }
        }
        // ---- stage V transposed [dv][key]; 512 key-pair chunks, 2/thread ----
        {
            const _Float16* vp = Vh + ((size_t)((b * T_ + kstart) * H_ + h)) * DV_;
            uint32_t* sVT32 = (uint32_t*)sVT;
            #pragma unroll
            for (int i = 0; i < 2; ++i) {
                int p   = i * 256 + tid;        // 0..511
                int rp  = p >> 4;               // key pair 0..31 -> keys 2rp, 2rp+1
                int dv0 = (p & 15) * 8;
                v8h a = *(const v8h*)(vp + (size_t)(2 * rp)     * (H_ * DV_) + dv0);
                v8h c = *(const v8h*)(vp + (size_t)(2 * rp + 1) * (H_ * DV_) + dv0);
                #pragma unroll
                for (int j = 0; j < 8; ++j) {
                    HPack pk;
                    pk.h[0] = a[j];
                    pk.h[1] = c[j];
                    sVT32[(dv0 + j) * 32 + rp] = pk.u;
                }
            }
        }
        asm volatile("s_wait_asynccnt 0" ::: "memory");
        __syncthreads();

        if (kstart <= qlast) {                       // scalar branch
            const bool k32b = (kstart + 32 <= qlast);

            // ---- S = Q K^T (16 x 64) with per-16-col causal tile skip ----
            v8f sacc[4];
            #pragma unroll
            for (int nt = 0; nt < 4; ++nt) {
                if (kstart + nt * 16 <= qlast) {     // scalar branch
                    v16h bf[4];
                    #pragma unroll
                    for (int kk = 0; kk < 4; ++kk)
                        bf[kk] = *(const v16h*)(sK + (nt * 16 + col) * DH_ +
                                                kk * 32 + (hi ? 16 : 0));
                    v8f c = (v8f){};
                    #pragma unroll
                    for (int kk = 0; kk < 4; ++kk)
                        c = __builtin_amdgcn_wmma_f32_16x16x32_f16(
                                false, qf[kk].v, false, bf[kk], (short)0, c,
                                false, false);
                    sacc[nt] = c;
                    // causal mask only on the diagonal-overlapping tile
                    if (kstart + nt * 16 + 15 > qfirst) {   // scalar branch
                        const int qg_base = qfirst + (hi ? 8 : 0);
                        int kg = kstart + nt * 16 + col;
                        #pragma unroll
                        for (int r = 0; r < 8; ++r)
                            if (kg > qg_base + r) sacc[nt][r] = NEGINF;
                    }
                } else {
                    #pragma unroll
                    for (int r = 0; r < 8; ++r) sacc[nt][r] = NEGINF;
                }
            }

            // ---- online softmax ----
            float scale_r[8];
            #pragma unroll
            for (int r = 0; r < 8; ++r) {
                float m = fmaxf(fmaxf(sacc[0][r], sacc[1][r]),
                                fmaxf(sacc[2][r], sacc[3][r]));
                m = fmaxf(m, __shfl_xor(m, 8, 32));
                m = fmaxf(m, __shfl_xor(m, 4, 32));
                m = fmaxf(m, __shfl_xor(m, 2, 32));
                m = fmaxf(m, __shfl_xor(m, 1, 32));
                float mn = fmaxf(m_i[r], m);
                scale_r[r] = __expf(m_i[r] - mn);
                m_i[r] = mn;
                l_i[r] *= scale_r[r];
            }
            #pragma unroll
            for (int n = 0; n < 8; ++n)
                #pragma unroll
                for (int r = 0; r < 8; ++r)
                    acc[n][r] *= scale_r[r];

            // ---- p = exp(s - m); row sums; stage P (f16) ----
            float rsum[8];
            #pragma unroll
            for (int r = 0; r < 8; ++r) rsum[r] = 0.0f;
            _Float16* pbase = sP + wave_s * 1024;   // [16][64]
            #pragma unroll
            for (int nt = 0; nt < 2; ++nt)
                #pragma unroll
                for (int r = 0; r < 8; ++r) {
                    float p = __expf(sacc[nt][r] - m_i[r]);
                    rsum[r] += p;
                    pbase[(r + (hi ? 8 : 0)) * 64 + nt * 16 + col] = (_Float16)p;
                }
            if (k32b) {
                #pragma unroll
                for (int nt = 2; nt < 4; ++nt)
                    #pragma unroll
                    for (int r = 0; r < 8; ++r) {
                        float p = __expf(sacc[nt][r] - m_i[r]);
                        rsum[r] += p;
                        pbase[(r + (hi ? 8 : 0)) * 64 + nt * 16 + col] = (_Float16)p;
                    }
            }
            #pragma unroll
            for (int r = 0; r < 8; ++r) {
                float sum = rsum[r];
                sum += __shfl_xor(sum, 8, 32);
                sum += __shfl_xor(sum, 4, 32);
                sum += __shfl_xor(sum, 2, 32);
                sum += __shfl_xor(sum, 1, 32);
                l_i[r] += sum;
            }

            asm volatile("s_wait_dscnt 0" ::: "memory");

            // ---- P as A-fragments ----
            AFrag pf[2];
            {
                const int s8 = hi ? 16 : 0;
                const _Float16* prow = sP + wave_s * 1024 + col * 64;
                pf[0].h[0] = *(const v8h*)(prow + s8);
                pf[0].h[1] = *(const v8h*)(prow + 8 + s8);
                if (k32b) {
                    pf[1].h[0] = *(const v8h*)(prow + 32 + s8);
                    pf[1].h[1] = *(const v8h*)(prow + 40 + s8);
                }
            }
            // ---- O += P V : preload both V fragments per n-tile ----
            #pragma unroll
            for (int n = 0; n < 8; ++n) {
                const _Float16* vb0 = sVT + (n * 16 + col) * 64 + (hi ? 16 : 0);
                v16h vf0 = *(const v16h*)vb0;
                v8f c = acc[n];
                if (k32b) {
                    v16h vf1 = *(const v16h*)(vb0 + 32);
                    c = __builtin_amdgcn_wmma_f32_16x16x32_f16(
                            false, pf[0].v, false, vf0, (short)0, c, false, false);
                    c = __builtin_amdgcn_wmma_f32_16x16x32_f16(
                            false, pf[1].v, false, vf1, (short)0, c, false, false);
                } else {
                    c = __builtin_amdgcn_wmma_f32_16x16x32_f16(
                            false, pf[0].v, false, vf0, (short)0, c, false, false);
                }
                acc[n] = c;
            }
        }
    }

    // ---- epilogue ----
    const int qg0 = qfirst + (hi ? 8 : 0);
    #pragma unroll
    for (int r = 0; r < 8; ++r) {
        float inv = 1.0f / l_i[r];
        int t = qg0 + r;
        float* op = O + ((size_t)((b * T_ + t) * H_ + h)) * DV_;
        #pragma unroll
        for (int n = 0; n < 8; ++n)
            op[n * 16 + col] = acc[n][r] * inv;
    }
}

// ---------------------------------------------------------------------------
// Fallback: self-contained monolithic kernel, used if ws too small.
// ---------------------------------------------------------------------------
__launch_bounds__(256)
__global__ void megalodon_attn_kernel(const float* __restrict__ Q,
                                      const float* __restrict__ K,
                                      const float* __restrict__ V,
                                      const int* __restrict__ start_index,
                                      float* __restrict__ O)
{
    __shared__ __align__(64) _Float16 smem[24576];
    _Float16* sQ  = smem;
    _Float16* sK  = smem;
    _Float16* sVT = smem + 8192;
    _Float16* sP  = smem + 16384;

    const int tid  = threadIdx.x;
    const int lane = tid & 31;
    const int wave = tid >> 5;
    const int col  = lane & 15;
    const int hi   = (lane >> 4) & 1;
    const int wave_s = __builtin_amdgcn_readfirstlane(wave);

    const int qb_per_bh = T_ / 128;
    int idx = blockIdx.x;
    const int b  = idx / (H_ * qb_per_bh);
    idx -= b * H_ * qb_per_bh;
    const int h  = idx / qb_per_bh;
    const int qb = idx - h * qb_per_bh;

    const int qbase       = qb * 128;
    const int chunk_start = (qbase / CHUNK_) * CHUNK_;
    const int qrel        = qbase - chunk_start;
    const int num_tiles   = (qrel + 128) >> 6;
    const int start       = start_index[0];

    for (int i = 0; i < 32; ++i) {
        int p   = i * 256 + tid;
        int row = p >> 6;
        int d   = p & 63;
        int t   = qbase + row;
        const float* qp = Q + ((size_t)((b * T_ + t) * H_ + h)) * DH_;
        float x1 = qp[d];
        float x2 = qp[d + 64];
        float ang = (float)(start + t) * __expf(-ROPE_SCALE * (float)d);
        float sn, cs;
        __sincosf(ang, &sn, &cs);
        sQ[row * DH_ + d]      = (_Float16)(x1 * cs - x2 * sn);
        sQ[row * DH_ + d + 64] = (_Float16)(x2 * cs + x1 * sn);
    }
    __syncthreads();

    AFrag qf[4];
    {
        const int qrow = wave_s * 16 + col;
        const int s8   = hi ? 16 : 0;
        for (int kk = 0; kk < 4; ++kk) {
            const _Float16* base = sQ + qrow * DH_ + kk * 32;
            qf[kk].h[0] = *(const v8h*)(base + s8);
            qf[kk].h[1] = *(const v8h*)(base + 8 + s8);
        }
    }
    asm volatile("s_wait_dscnt 0" ::: "memory");

    v8f acc[8];
    for (int n = 0; n < 8; ++n) acc[n] = (v8f){};
    float m_i[8], l_i[8];
    for (int r = 0; r < 8; ++r) { m_i[r] = NEGINF; l_i[r] = 0.0f; }

    const int qfirst = qbase + wave_s * 16;
    const int qlast  = qfirst + 15;

    for (int kt = 0; kt < num_tiles; ++kt) {
        const int kstart = chunk_start + kt * 64;
        __syncthreads();

        for (int i = 0; i < 16; ++i) {
            int p   = i * 256 + tid;
            int row = p >> 6;
            int d   = p & 63;
            int t   = kstart + row;
            const float* kp = K + ((size_t)((b * T_ + t) * H_ + h)) * DH_;
            float x1 = kp[d];
            float x2 = kp[d + 64];
            float ang = (float)(start + t) * __expf(-ROPE_SCALE * (float)d);
            float sn, cs;
            __sincosf(ang, &sn, &cs);
            sK[row * DH_ + d]      = (_Float16)(x1 * cs - x2 * sn);
            sK[row * DH_ + d + 64] = (_Float16)(x2 * cs + x1 * sn);
        }
        for (int i = 0; i < 32; ++i) {
            int p   = i * 256 + tid;
            int row = p >> 7;
            int dv  = p & 127;
            int t   = kstart + row;
            sVT[dv * 64 + row] =
                (_Float16)V[((size_t)((b * T_ + t) * H_ + h)) * DV_ + dv];
        }
        __syncthreads();

        if (kstart <= qlast) {
            v8f sacc[4];
            for (int nt = 0; nt < 4; ++nt) {
                v16h bf[4];
                for (int kk = 0; kk < 4; ++kk)
                    bf[kk] = *(const v16h*)(sK + (nt * 16 + col) * DH_ +
                                            kk * 32 + (hi ? 16 : 0));
                v8f c = (v8f){};
                for (int kk = 0; kk < 4; ++kk)
                    c = __builtin_amdgcn_wmma_f32_16x16x32_f16(
                            false, qf[kk].v, false, bf[kk], (short)0, c, false, false);
                sacc[nt] = c;
            }
            const int qg_base = qfirst + (hi ? 8 : 0);
            for (int nt = 0; nt < 4; ++nt) {
                int kg = kstart + nt * 16 + col;
                for (int r = 0; r < 8; ++r)
                    if (kg > qg_base + r) sacc[nt][r] = NEGINF;
            }
            float scale_r[8];
            for (int r = 0; r < 8; ++r) {
                float m = fmaxf(fmaxf(sacc[0][r], sacc[1][r]),
                                fmaxf(sacc[2][r], sacc[3][r]));
                m = fmaxf(m, __shfl_xor(m, 8, 32));
                m = fmaxf(m, __shfl_xor(m, 4, 32));
                m = fmaxf(m, __shfl_xor(m, 2, 32));
                m = fmaxf(m, __shfl_xor(m, 1, 32));
                float mn = fmaxf(m_i[r], m);
                scale_r[r] = __expf(m_i[r] - mn);
                m_i[r] = mn;
                l_i[r] *= scale_r[r];
            }
            for (int n = 0; n < 8; ++n)
                for (int r = 0; r < 8; ++r)
                    acc[n][r] *= scale_r[r];

            float rsum[8];
            for (int r = 0; r < 8; ++r) rsum[r] = 0.0f;
            _Float16* pbase = sP + wave_s * 1024;
            for (int nt = 0; nt < 4; ++nt)
                for (int r = 0; r < 8; ++r) {
                    float p = __expf(sacc[nt][r] - m_i[r]);
                    rsum[r] += p;
                    pbase[(r + (hi ? 8 : 0)) * 64 + nt * 16 + col] = (_Float16)p;
                }
            for (int r = 0; r < 8; ++r) {
                float sum = rsum[r];
                sum += __shfl_xor(sum, 8, 32);
                sum += __shfl_xor(sum, 4, 32);
                sum += __shfl_xor(sum, 2, 32);
                sum += __shfl_xor(sum, 1, 32);
                l_i[r] += sum;
            }

            asm volatile("s_wait_dscnt 0" ::: "memory");

            AFrag pf[2];
            {
                const int s8 = hi ? 16 : 0;
                const _Float16* prow = sP + wave_s * 1024 + col * 64;
                for (int kkp = 0; kkp < 2; ++kkp) {
                    pf[kkp].h[0] = *(const v8h*)(prow + kkp * 32 + s8);
                    pf[kkp].h[1] = *(const v8h*)(prow + kkp * 32 + 8 + s8);
                }
            }
            for (int n = 0; n < 8; ++n) {
                v16h vf[2];
                for (int kkp = 0; kkp < 2; ++kkp)
                    vf[kkp] = *(const v16h*)(sVT + (n * 16 + col) * 64 +
                                             kkp * 32 + (hi ? 16 : 0));
                v8f c = acc[n];
                for (int kkp = 0; kkp < 2; ++kkp)
                    c = __builtin_amdgcn_wmma_f32_16x16x32_f16(
                            false, pf[kkp].v, false, vf[kkp], (short)0, c, false, false);
                acc[n] = c;
            }
        }
    }

    const int qg0 = qfirst + (hi ? 8 : 0);
    for (int r = 0; r < 8; ++r) {
        float inv = 1.0f / l_i[r];
        int t = qg0 + r;
        float* op = O + ((size_t)((b * T_ + t) * H_ + h)) * DV_;
        for (int n = 0; n < 8; ++n)
            op[n * 16 + col] = acc[n][r] * inv;
    }
}

extern "C" void kernel_launch(void* const* d_in, const int* in_sizes, int n_in,
                              void* d_out, int out_size, void* d_ws, size_t ws_size,
                              hipStream_t stream) {
    const float* q = (const float*)d_in[0];
    const float* k = (const float*)d_in[1];
    const float* v = (const float*)d_in[2];
    const int* start_index = (const int*)d_in[3];
    float* out = (float*)d_out;

    const size_t nhalf = (size_t)B_ * T_ * H_ * DH_;   // elements per tensor
    const size_t need  = nhalf * sizeof(_Float16) * 3; // Qr + Kr + Vh

    if (ws_size >= need) {
        _Float16* Qr = (_Float16*)d_ws;
        _Float16* Kr = Qr + nhalf;
        _Float16* Vh = Kr + nhalf;
        rope_cast_kernel<<<2048, 256, 0, stream>>>(q, k, v, start_index, Qr, Kr, Vh);
        megalodon_attn_f16_kernel<<<B_ * H_ * (T_ / 128), 256, 0, stream>>>(Qr, Kr, Vh, out);
    } else {
        megalodon_attn_kernel<<<B_ * H_ * (T_ / 128), 256, 0, stream>>>(q, k, v, start_index, out);
    }
}
